// UncertainTemporalAttention_68693706932587
// MI455X (gfx1250) — matrix operand order repaired and checked by reading, not compiled, not run on hardware
//
#include <hip/hip_runtime.h>

#define N_ROWS 8192
#define DIM    256
#define KTILE  32
#define QBLK   64   // 4 waves x 16 rows

typedef __attribute__((ext_vector_type(16))) __bf16 v16bf;
typedef __attribute__((ext_vector_type(8)))  __bf16 v8bf;
typedef __attribute__((ext_vector_type(8)))  float  v8f;

__device__ inline v8f wmma_bf16(v16bf a, v16bf b, v8f c) {
  return __builtin_amdgcn_wmma_f32_16x16x32_bf16(
      /*neg_a=*/false, a, /*neg_b=*/false, b,
      /*c_mod=*/(short)0, c, /*reuse_a=*/false, /*reuse_b=*/false);
}

// 16 contiguous bf16 -> one B fragment (two 16B loads)
__device__ inline v16bf ld16(const __bf16* p) {
  union { v16bf v; v8bf h[2]; } u;
  u.h[0] = *(const v8bf*)(p);
  u.h[1] = *(const v8bf*)(p + 8);
  return u.v;
}

// A-fragment: two 8-element runs at p and p+16 (K-gap of 8 per ISA layout)
__device__ inline v16bf ldA(const __bf16* p) {
  union { v16bf v; v8bf h[2]; } u;
  u.h[0] = *(const v8bf*)(p);
  u.h[1] = *(const v8bf*)(p + 16);
  return u.v;
}

// 32-bit offset of a generic pointer within the workgroup LDS segment
__device__ inline uint32_t lds_off32(const void* p) {
  return (uint32_t)(uintptr_t)(__attribute__((address_space(3))) const void*)p;
}

// Async DMA: 16 bytes global -> LDS, tracked by ASYNCcnt.
// INST_OFFSET is applied to both the LDS and the global address (ISA 08 §4.4),
// so equal-stride chunk sequences can share one base register pair.
template <int OFF>
__device__ inline void async_copy16(uint32_t lds_byte_off, const void* g) {
  asm volatile("global_load_async_to_lds_b128 %0, %1, off offset:%2"
               :: "v"(lds_byte_off), "v"((unsigned long long)(uintptr_t)g), "i"(OFF)
               : "memory");
}

// ---------------------------------------------------------------------------
// Kernel 0: one-time fp32 -> bf16 conversion (inputs and weights)
// ---------------------------------------------------------------------------
__global__ __launch_bounds__(256) void cvt_f32_bf16(const float* __restrict__ src,
                                                    __bf16* __restrict__ dst, int n) {
  const int i = (blockIdx.x * 256 + threadIdx.x) * 8;
  if (i + 8 <= n) {
    float4 a = *(const float4*)(src + i);
    float4 b = *(const float4*)(src + i + 4);
    v8bf o;
    o[0] = (__bf16)a.x; o[1] = (__bf16)a.y; o[2] = (__bf16)a.z; o[3] = (__bf16)a.w;
    o[4] = (__bf16)b.x; o[5] = (__bf16)b.y; o[6] = (__bf16)b.z; o[7] = (__bf16)b.w;
    *(v8bf*)(dst + i) = o;
  }
}

// ---------------------------------------------------------------------------
// Kernel 1/3: Y = X @ W^T + b, all-bf16 inputs. Wave = 16 rows x 64 cols
// (4 WMMA tiles sharing one A fragment). Block = 4 waves = full 256 cols.
// ---------------------------------------------------------------------------
template <bool OUT_F32, bool TRANS>
__global__ __launch_bounds__(128) void proj64(const __bf16* __restrict__ X,
                                              const __bf16* __restrict__ W,
                                              const float* __restrict__ Bias,
                                              void* __restrict__ Yout) {
  const int wave = threadIdx.x >> 5;
  const int lane = threadIdx.x & 31;
  const int half = lane >> 4;
  const int ln   = lane & 15;
  const int rbase = blockIdx.x * 16;
  const int dgrp  = wave * 64;

  v8f acc[4] = {};
#pragma unroll
  for (int kb = 0; kb < 8; ++kb) {
    v16bf a = ld16(X + (size_t)(rbase + ln) * DIM + kb * 32 + half * 16);
#pragma unroll
    for (int t = 0; t < 4; ++t) {
      v16bf b = ld16(W + (size_t)(dgrp + t * 16 + ln) * DIM + kb * 32 + half * 16);
      acc[t] = wmma_bf16(a, b, acc[t]);
    }
  }

#pragma unroll
  for (int t = 0; t < 4; ++t) {
    const int col = dgrp + t * 16 + ln;
    const float bias = Bias[col];
    if (TRANS) {
      v8bf pk;
#pragma unroll
      for (int r = 0; r < 8; ++r) pk[r] = (__bf16)(acc[t][r] + bias);
      *(v8bf*)((__bf16*)Yout + (size_t)col * N_ROWS + rbase + half * 8) = pk;
    } else {
#pragma unroll
      for (int r = 0; r < 8; ++r) {
        const size_t idx = (size_t)(rbase + r + half * 8) * DIM + col;
        if (OUT_F32) ((float*)Yout)[idx] = acc[t][r] + bias;
        else         ((__bf16*)Yout)[idx] = (__bf16)(acc[t][r] + bias);
      }
    }
  }
}

// ---------------------------------------------------------------------------
// Kernel 2: fused flash attention with variance path, double-buffered async
// DMA staging (DMA for tile t+1 overlaps WMMA compute on tile t).
//   AO     = attn @ V            (bf16 row-major, pre out-projection)
//   OutVar = (attn*attn) @ Vvar  (fp32, final output half 2)
// ---------------------------------------------------------------------------
__global__ __launch_bounds__(128) void flash_attn(const __bf16* __restrict__ Q,
                                                  const __bf16* __restrict__ K,
                                                  const __bf16* __restrict__ Vt,
                                                  const __bf16* __restrict__ VVt,
                                                  __bf16* __restrict__ AO,
                                                  float* __restrict__ OutVar) {
  __shared__ __align__(16) __bf16 sK[2][KTILE * DIM];    // [key][feat] row-major
  __shared__ __align__(16) __bf16 sVt[2][DIM * KTILE];   // [feat][key]
  __shared__ __align__(16) __bf16 sVVt[2][DIM * KTILE];  // [feat][key]
  __shared__ __align__(16) __bf16 sP[4][16 * KTILE];     // per-wave P scratch

  const int tid  = threadIdx.x;
  const int wave = tid >> 5;
  const int lane = tid & 31;
  const int half = lane >> 4;
  const int ln   = lane & 15;
  const int qrow = blockIdx.x * QBLK + wave * 16;
  const float scale = 0.125f;  // head_dim=64 -> 1/sqrt(64)

  const uint32_t sK_base   = lds_off32(sK);
  const uint32_t sVt_base  = lds_off32(sVt);
  const uint32_t sVVt_base = lds_off32(sVVt);

  // Per-thread staging invariants (hoisted; loop just advances base pointers)
  const uint32_t kOff = (uint32_t)tid * 16;            // K tile: +i*2048 (both)
  const int      d0   = tid >> 2;                      // V rows: d0 + 32*i
  const uint32_t c0   = (uint32_t)(tid & 3) * 16;      // byte chunk in 64B row
  const char* aK  = (const char*)K + kOff;             // advances 16KB / tile
  const char* aV  = (const char*)Vt  + (size_t)d0 * (N_ROWS * 2) + c0;  // +64B / tile
  const char* aVV = (const char*)VVt + (size_t)d0 * (N_ROWS * 2) + c0;  // +64B / tile
  const uint32_t lK0 = sK_base + kOff;
  const uint32_t lV0 = sVt_base + (uint32_t)d0 * 64 + c0;
  const uint32_t lVV0 = sVVt_base + (uint32_t)d0 * 64 + c0;

  auto stage = [&](int kb, int par) {
    const uint32_t pb = (uint32_t)par * (KTILE * DIM * 2);
    const char* gk = aK + (size_t)kb * (DIM * 2);
#pragma unroll
    for (int i = 0; i < 8; ++i)
      async_copy16<0>(lK0 + pb + i * 2048, gk + i * 2048);
    const char* gv  = aV  + (size_t)kb * 2;
    const char* gvv = aVV + (size_t)kb * 2;
#pragma unroll
    for (int i = 0; i < 8; ++i) {
      async_copy16<0>(lV0 + pb + i * 2048, gv + (size_t)i * 32 * (N_ROWS * 2));
      async_copy16<0>(lVV0 + pb + i * 2048, gvv + (size_t)i * 32 * (N_ROWS * 2));
    }
  };

  // Kick off DMA for tile 0, then overlap the Q-fragment preload with it.
  stage(0, 0);

  v16bf qf[8];
#pragma unroll
  for (int kb = 0; kb < 8; ++kb)
    qf[kb] = ld16(Q + (size_t)(qrow + ln) * DIM + kb * 32 + half * 16);

  float m_run[8], l_run[8];
  v8f accM[16], accV[16];
#pragma unroll
  for (int r = 0; r < 8; ++r) { m_run[r] = -1e30f; l_run[r] = 0.0f; }
#pragma unroll
  for (int dt = 0; dt < 16; ++dt) { accM[dt] = (v8f){}; accV[dt] = (v8f){}; }

  int parity = 0;
  for (int kbase = 0; kbase < N_ROWS; kbase += KTILE) {
    // issue DMA for next tile into the other buffer, drain current batch
    if (kbase + KTILE < N_ROWS) {
      stage(kbase + KTILE, parity ^ 1);
      asm volatile("s_wait_asynccnt 0x18" ::: "memory");  // <= one batch (24)
    } else {
      asm volatile("s_wait_asynccnt 0x0" ::: "memory");
    }
    __syncthreads();

    const __bf16* cK  = sK[parity];
    const __bf16* cVt = sVt[parity];
    const __bf16* cVV = sVVt[parity];

    // ---- S = Q @ K^T for 32 keys (two 16x16 C tiles) ----
    v8f s0 = {}, s1 = {};
#pragma unroll
    for (int kb = 0; kb < 8; ++kb) {
      v16bf b0 = ld16(cK + (size_t)ln * DIM + kb * 32 + half * 16);
      v16bf b1 = ld16(cK + (size_t)(16 + ln) * DIM + kb * 32 + half * 16);
      s0 = wmma_bf16(qf[kb], b0, s0);
      s1 = wmma_bf16(qf[kb], b1, s1);
    }

    // ---- online softmax update (rows live across lanes 0..15 per half) ----
    float p0[8], p1[8];
#pragma unroll
    for (int r = 0; r < 8; ++r) {
      const float sc0 = s0[r] * scale, sc1 = s1[r] * scale;
      float mx = fmaxf(sc0, sc1);
      mx = fmaxf(mx, __shfl_xor(mx, 1));
      mx = fmaxf(mx, __shfl_xor(mx, 2));
      mx = fmaxf(mx, __shfl_xor(mx, 4));
      mx = fmaxf(mx, __shfl_xor(mx, 8));
      const float mnew = fmaxf(m_run[r], mx);
      const float corr = __expf(m_run[r] - mnew);
      const float e0 = __expf(sc0 - mnew), e1 = __expf(sc1 - mnew);
      float rs = e0 + e1;
      rs += __shfl_xor(rs, 1);
      rs += __shfl_xor(rs, 2);
      rs += __shfl_xor(rs, 4);
      rs += __shfl_xor(rs, 8);
      l_run[r] = l_run[r] * corr + rs;
      m_run[r] = mnew;
      p0[r] = e0; p1[r] = e1;
      const float c2 = corr * corr;
#pragma unroll
      for (int dt = 0; dt < 16; ++dt) { accM[dt][r] *= corr; accV[dt][r] *= c2; }
    }

    // ---- P through per-wave LDS scratch: C-layout -> A-fragment ----
    __bf16* pw = sP[wave];
#pragma unroll
    for (int r = 0; r < 8; ++r) {
      const int row = r + half * 8;
      pw[row * KTILE + ln]      = (__bf16)p0[r];
      pw[row * KTILE + 16 + ln] = (__bf16)p1[r];
    }
    asm volatile("s_wait_dscnt 0" ::: "memory");
    v16bf pf = ld16(pw + ln * KTILE + half * 16);
#pragma unroll
    for (int dt = 0; dt < 16; ++dt) {
      v16bf bv = ld16(cVt + (size_t)(dt * 16 + ln) * KTILE + half * 16);
      accM[dt] = wmma_bf16(pf, bv, accM[dt]);
    }

    // ---- P^2 for the variance path ----
#pragma unroll
    for (int r = 0; r < 8; ++r) {
      const int row = r + half * 8;
      pw[row * KTILE + ln]      = (__bf16)(p0[r] * p0[r]);
      pw[row * KTILE + 16 + ln] = (__bf16)(p1[r] * p1[r]);
    }
    asm volatile("s_wait_dscnt 0" ::: "memory");
    v16bf p2f = ld16(pw + ln * KTILE + half * 16);
#pragma unroll
    for (int dt = 0; dt < 16; ++dt) {
      v16bf bv = ld16(cVV + (size_t)(dt * 16 + ln) * KTILE + half * 16);
      accV[dt] = wmma_bf16(p2f, bv, accV[dt]);
    }
    __syncthreads();
    parity ^= 1;
  }

  // ---- finalize: mean /= l (bf16 to AO), var /= l^2 (fp32 to output) ----
#pragma unroll
  for (int r = 0; r < 8; ++r) {
    const float inv  = 1.0f / l_run[r];
    const float inv2 = inv * inv;
    const size_t row = (size_t)(qrow + r + half * 8) * DIM;
#pragma unroll
    for (int dt = 0; dt < 16; ++dt) {
      AO[row + dt * 16 + ln]     = (__bf16)(accM[dt][r] * inv);
      OutVar[row + dt * 16 + ln] = accV[dt][r] * inv2;
    }
  }
}

// ---------------------------------------------------------------------------
extern "C" void kernel_launch(void* const* d_in, const int* in_sizes, int n_in,
                              void* d_out, int out_size, void* d_ws, size_t ws_size,
                              hipStream_t stream) {
  const float* xm = (const float*)d_in[0];
  const float* xv = (const float*)d_in[1];
  // d_in[2] edge_index, d_in[3] edge_timestamps: unused by the reference math
  const float* Wq = (const float*)d_in[4],  *bq = (const float*)d_in[5];
  const float* Wk = (const float*)d_in[6],  *bk = (const float*)d_in[7];
  const float* Wv = (const float*)d_in[8],  *bv = (const float*)d_in[9];
  const float* Wo = (const float*)d_in[10], *bo = (const float*)d_in[11];
  const float* Wvar = (const float*)d_in[12], *bvar = (const float*)d_in[13];

  float* out_mean = (float*)d_out;
  float* out_var  = out_mean + (size_t)N_ROWS * DIM;

  const size_t nd = (size_t)N_ROWS * DIM;   // 2M elems
  const size_t dd = (size_t)DIM * DIM;      // 64K elems
  char* ws = (char*)d_ws;
  size_t o = 0;
  __bf16* Xmb  = (__bf16*)(ws + o); o += nd * 2;
  __bf16* Xvb  = (__bf16*)(ws + o); o += nd * 2;
  __bf16* Wqb  = (__bf16*)(ws + o); o += dd * 2;
  __bf16* Wkb  = (__bf16*)(ws + o); o += dd * 2;
  __bf16* Wvb  = (__bf16*)(ws + o); o += dd * 2;
  __bf16* Wvrb = (__bf16*)(ws + o); o += dd * 2;
  __bf16* Wob  = (__bf16*)(ws + o); o += dd * 2;
  __bf16* Qb   = (__bf16*)(ws + o); o += nd * 2;   // [row][d]
  __bf16* Kb   = (__bf16*)(ws + o); o += nd * 2;   // [row][d]
  __bf16* Vtb  = (__bf16*)(ws + o); o += nd * 2;   // [d][row]
  __bf16* VVtb = (__bf16*)(ws + o); o += nd * 2;   // [d][row]
  __bf16* AOb  = (__bf16*)(ws + o); o += nd * 2;   // [row][d]

  // one-time bf16 conversion of activations and weights
  cvt_f32_bf16<<<dim3((unsigned)(nd / 8 / 256)), 256, 0, stream>>>(xm, Xmb, (int)nd);
  cvt_f32_bf16<<<dim3((unsigned)(nd / 8 / 256)), 256, 0, stream>>>(xv, Xvb, (int)nd);
  cvt_f32_bf16<<<dim3((unsigned)(dd / 8 / 256)), 256, 0, stream>>>(Wq, Wqb, (int)dd);
  cvt_f32_bf16<<<dim3((unsigned)(dd / 8 / 256)), 256, 0, stream>>>(Wk, Wkb, (int)dd);
  cvt_f32_bf16<<<dim3((unsigned)(dd / 8 / 256)), 256, 0, stream>>>(Wv, Wvb, (int)dd);
  cvt_f32_bf16<<<dim3((unsigned)(dd / 8 / 256)), 256, 0, stream>>>(Wvar, Wvrb, (int)dd);
  cvt_f32_bf16<<<dim3((unsigned)(dd / 8 / 256)), 256, 0, stream>>>(Wo, Wob, (int)dd);

  dim3 pgrid(N_ROWS / 16);   // 512 blocks x 4 waves = 16-row stripe x 256 cols
  proj64<false, false><<<pgrid, 128, 0, stream>>>(Xmb, Wqb, bq, (void*)Qb);
  proj64<false, false><<<pgrid, 128, 0, stream>>>(Xmb, Wkb, bk, (void*)Kb);
  proj64<false, true ><<<pgrid, 128, 0, stream>>>(Xmb, Wvb, bv, (void*)Vtb);
  proj64<false, true ><<<pgrid, 128, 0, stream>>>(Xvb, Wvrb, bvar, (void*)VVtb);

  flash_attn<<<dim3(N_ROWS / QBLK), 128, 0, stream>>>(Qb, Kb, Vtb, VVtb, AOb, out_var);

  proj64<true, false><<<pgrid, 128, 0, stream>>>(AOb, Wob, bo, (void*)out_mean);
}
